// PostIoU_5566277615935
// MI455X (gfx1250) — compile-verified, matrix operand order checked by reading, not run on hardware
//
#include <hip/hip_runtime.h>
#include <hip/hip_bf16.h>

typedef __attribute__((ext_vector_type(16))) _Float16 v16h;
typedef __attribute__((ext_vector_type(8)))  float    v8f;

#define GLOBAL_AS __attribute__((address_space(1)))
#define LDS_AS    __attribute__((address_space(3)))

#if __has_builtin(__builtin_amdgcn_global_load_async_to_lds_b32)
#define USE_ASYNC_LDS 1
#else
#define USE_ASYNC_LDS 0
#endif

// ---------------------------------------------------------------------------
// Zero the per-batch counters (2 x 32 unsigned) in workspace each call.
// ---------------------------------------------------------------------------
__global__ void postiou_zero(unsigned* __restrict__ cnt) {
    if (threadIdx.x < 64) cnt[threadIdx.x] = 0u;
}

// ---------------------------------------------------------------------------
// Fused argmax + 3x3 erode/dilate + per-batch inter/uni counting.
// Grid: (W/16, H/16, B), block: 256 threads (16x16 pixel tile, 8 waves).
// ---------------------------------------------------------------------------
__global__ __launch_bounds__(256) void postiou_main(const float* __restrict__ yhat,
                                                    const int*   __restrict__ y,
                                                    unsigned*    __restrict__ cnt)
{
    constexpr int H = 512, W = 512, HW = H * W;
    __shared__ int tile[18 * 18];

    const int b   = blockIdx.z;
    const int tid = threadIdx.x;
    const int ty  = tid >> 4, tx = tid & 15;
    const int h   = blockIdx.y * 16 + ty;
    const int w   = blockIdx.x * 16 + tx;
    const int h0  = (int)blockIdx.y * 16 - 1;
    const int w0  = (int)blockIdx.x * 16 - 1;
    const int* ybase = y + (size_t)b * HW;

    // --- stage the 18x18 y-halo tile into LDS (addresses clamped in-bounds;
    //     OOB semantics handled later from coordinates, not data) ---
    for (int t = tid; t < 324; t += 256) {
        int r  = t / 18, c = t - r * 18;
        int gy = min(max(h0 + r, 0), H - 1);
        int gx = min(max(w0 + c, 0), W - 1);
        const int* gp = ybase + gy * W + gx;
#if USE_ASYNC_LDS
        __builtin_amdgcn_global_load_async_to_lds_b32(
            (GLOBAL_AS int*)(const_cast<int*>(gp)),
            (LDS_AS int*)&tile[t], /*offset=*/0, /*cpol=*/0);
#else
        tile[t] = *gp;
#endif
    }
#if USE_ASYNC_LDS
#if __has_builtin(__builtin_amdgcn_s_wait_asynccnt)
    __builtin_amdgcn_s_wait_asynccnt(0);
#else
    asm volatile("s_wait_asynccnt 0" ::: "memory");
#endif
#endif
    __syncthreads();

    // --- argmax over 3 channels (first-max tie-break, matches jnp.argmax) ---
    const float* cb = yhat + (size_t)b * 3 * HW + (size_t)h * W + w;
    float f0 = cb[0], f1 = cb[HW], f2 = cb[2 * HW];
    int   p = 0; float best = f0;
    if (f1 > best) { best = f1; p = 1; }
    if (f2 > best) { p = 2; }

    // --- 3x3 erode (AND) / dilate (OR); OOB neighbors are AND-neutral /
    //     OR-neutral, matching reduce_window's +/-inf identities ---
    unsigned andv = 3u, orv = 0u;
#pragma unroll
    for (int dy = -1; dy <= 1; ++dy) {
#pragma unroll
        for (int dx = -1; dx <= 1; ++dx) {
            int  hh = h + dy, ww = w + dx;
            bool valid = ((unsigned)hh < (unsigned)H) && ((unsigned)ww < (unsigned)W);
            int  yv = tile[(ty + 1 + dy) * 18 + (tx + 1 + dx)];
            unsigned bits = (yv == 1 ? 1u : 0u) | (yv == 2 ? 2u : 0u);
            andv &= valid ? bits : 3u;
            orv  |= valid ? bits : 0u;
        }
    }
    int eroded  = (andv & 2u) ? 2 : ((andv & 1u) ? 1 : 0);
    int dilated = (orv  & 2u) ? 2 : ((orv  & 1u) ? 1 : 0);
    unsigned inter = (dilated != 0 && dilated == p) ? 1u : 0u;
    unsigned uni   = (eroded  != 0 || p       != 0) ? 1u : 0u;

    // --- per-wave reduction on the matrix pipe:
    // each lane broadcasts x = inter + 64*uni (<=65, exact in f16) into all 16
    // A slots; B = ones.  Per the 16-bit A layout, row m holds 16 copies of
    // x[m] and 16 of x[m+16], so D[m][n] = 16*(x[m]+x[m+16]).  Summing a
    // lane's 8 C regs and adding lane^16's sum yields 16*sum(all 32 lanes),
    // exactly representable in f32. ---
    _Float16 hv = (_Float16)(float)(inter + (uni << 6));
    v16h a, ones;
#pragma unroll
    for (int i = 0; i < 16; ++i) { a[i] = hv; ones[i] = (_Float16)1.0f; }
    v8f c = {};
    c = __builtin_amdgcn_wmma_f32_16x16x32_f16(
            /*neg_a=*/false, a, /*neg_b=*/false, ones,
            /*c_mod=*/(short)0, c, /*reuse_a=*/false, /*reuse_b=*/false);
    float s = ((c[0] + c[1]) + (c[2] + c[3])) + ((c[4] + c[5]) + (c[6] + c[7]));
    s += __shfl_xor(s, 16, 32);
    unsigned S = (unsigned)(s * (1.0f / 16.0f) + 0.5f);  // = sum(inter) + 64*sum(uni)
    if ((tid & 31) == 0) {
        atomicAdd(&cnt[2 * b + 0], S & 63u);
        atomicAdd(&cnt[2 * b + 1], S >> 6);
    }
}

// ---------------------------------------------------------------------------
// score = mean_b( inter[b] / uni[b] )
// ---------------------------------------------------------------------------
__global__ void postiou_final(const unsigned* __restrict__ cnt, float* __restrict__ out)
{
    int   b  = threadIdx.x;  // 32 threads, one wave
    float sc = (float)cnt[2 * b] / (float)cnt[2 * b + 1];
#pragma unroll
    for (int off = 16; off; off >>= 1) sc += __shfl_down(sc, off, 32);
    if (b == 0) out[0] = sc * (1.0f / 32.0f);
}

extern "C" void kernel_launch(void* const* d_in, const int* in_sizes, int n_in,
                              void* d_out, int out_size, void* d_ws, size_t ws_size,
                              hipStream_t stream) {
    const float* yhat = (const float*)d_in[0];   // [32,3,512,512] f32
    const int*   y    = (const int*)d_in[1];     // [32,512,512] i32
    float*       out  = (float*)d_out;           // scalar f32
    unsigned*    cnt  = (unsigned*)d_ws;         // [2*32] counters

    postiou_zero<<<1, 64, 0, stream>>>(cnt);
    dim3 grid(512 / 16, 512 / 16, 32);
    postiou_main<<<grid, 256, 0, stream>>>(yhat, y, cnt);
    postiou_final<<<1, 32, 0, stream>>>(cnt, out);
}
